// DGCNNCls_47236050321797
// MI455X (gfx1250) — matrix operand order repaired and checked by reading, not compiled, not run on hardware
//
#include <hip/hip_runtime.h>
#include <hip/hip_bf16.h>

// DGCNN classifier forward for gfx1250 (MI455X), wave32 + WMMA f16->f32.
// Feature maps live in zero-padded f16 buffers; WMMA fragments load as
// contiguous b128 chunks. GEMM-shaped kernels use 4x2 register blocking per
// wave (8 WMMAs per K-chunk, one loadcnt wait amortized over all 8).
typedef __attribute__((ext_vector_type(16))) _Float16 v16h;
typedef __attribute__((ext_vector_type(8)))  _Float16 v8h;
typedef __attribute__((ext_vector_type(8)))  float    v8f;

#define BQ   16      // batch
#define NP   1024    // points
#define KNN  20      // neighbors
#define SLOPE 0.2f
#define EPS   1e-5f

__device__ __forceinline__ float lrelu(float v) { return v >= 0.f ? v : SLOPE * v; }

// --- WMMA 16x16x32 f16 fragment loads (ISA 7.12.2, wave32) ------------------
// A (16x32 MxK): lane covers row M=lane&15; halves 0..7 -> K=8*hi+t,
// halves 8..15 -> K=16+8*hi+(t-8): two contiguous 16B runs.
__device__ __forceinline__ v16h ld_fragA(const _Float16* rowp, int hi) {
  v8h l = *(const v8h*)(rowp + 8 * hi);
  v8h h = *(const v8h*)(rowp + 16 + 8 * hi);
  v16h r;
#pragma unroll
  for (int i = 0; i < 8; ++i) { r[i] = l[i]; r[8 + i] = h[i]; }
  return r;
}
// B (32x16 KxN): lane covers col N=lane&15; halves 0..7 -> K=8*hi+t,
// halves 8..15 -> K=16+8*hi+(t-8): two contiguous 16B runs.
__device__ __forceinline__ v16h ld_fragB(const _Float16* colp, int hi) {
  v8h l = *(const v8h*)(colp + 8 * hi);
  v8h h = *(const v8h*)(colp + 16 + 8 * hi);
  v16h r;
#pragma unroll
  for (int i = 0; i < 8; ++i) { r[i] = l[i]; r[8 + i] = h[i]; }
  return r;
}

// ---------------------------------------------------------------- input prep
// x (B,3,N) f32 -> h0 (B*N, 32) f16, channels 3..31 zero.
__global__ void k_prep0(const float* __restrict__ x, _Float16* __restrict__ h0) {
  int t = blockIdx.x * blockDim.x + threadIdx.x;
  if (t >= BQ * NP) return;
  int b = t / NP, n = t % NP;
#pragma unroll
  for (int c = 0; c < 32; ++c)
    h0[(size_t)t * 32 + c] =
        (c < 3) ? (_Float16)x[((size_t)b * 3 + c) * NP + n] : (_Float16)0.f;
}

// w (O, Kin) f32 -> wp (O, Kpad) f16, zero padded.
__global__ void k_castw(const float* __restrict__ w, _Float16* __restrict__ wp,
                        int O, int Kin, int Kpad) {
  int t = blockIdx.x * blockDim.x + threadIdx.x;
  if (t >= O * Kpad) return;
  int o = t / Kpad, c = t % Kpad;
  wp[t] = (c < Kin) ? (_Float16)w[(size_t)o * Kin + c] : (_Float16)0.f;
}

// ---------------------------------------------------------------- sq norms (from f16, consistent with Gram)
__global__ void k_sqnorm(const _Float16* __restrict__ h, int strideF, int Cpad,
                         float* __restrict__ sq) {
  int t = blockIdx.x * blockDim.x + threadIdx.x;
  if (t >= BQ * NP) return;
  const _Float16* row = h + (size_t)t * strideF;
  float s = 0.f;
  for (int c = 0; c < Cpad; c += 8) {
    v8h v = *(const v8h*)(row + c);
#pragma unroll
    for (int i = 0; i < 8; ++i) { float f = (float)v[i]; s += f * f; }
  }
  sq[t] = s;
}

// ---------------------------------------------------------------- pd = 2*H H^T - sq_i - sq_j
// grid (NP/128, NP/128, B), block 256 (8 waves). Wave computes 64x32 (4 M-tiles
// x 2 N-tiles): 8 WMMAs per K-chunk off 6 fragment loads.
template <int CPAD>
__global__ void k_pd_wmma(const _Float16* __restrict__ h, int strideF,
                          const float* __restrict__ sq, float* __restrict__ pd) {
  int b = blockIdx.z;
  int wave = threadIdx.x >> 5, lane = threadIdx.x & 31;
  int lo = lane & 15, hi = lane >> 4;
  int mbase = blockIdx.x * 128 + (wave & 1) * 64;
  int nbase = blockIdx.y * 128 + (wave >> 1) * 32;
  const _Float16* hb = h + (size_t)b * NP * strideF;
  v8f acc[4][2] = {};
#pragma unroll
  for (int k0 = 0; k0 < CPAD; k0 += 32) {
    v16h a[4], bm[2];
#pragma unroll
    for (int i = 0; i < 4; ++i)
      a[i] = ld_fragA(hb + (size_t)(mbase + 16 * i + lo) * strideF + k0, hi);
#pragma unroll
    for (int j = 0; j < 2; ++j)
      bm[j] = ld_fragB(hb + (size_t)(nbase + 16 * j + lo) * strideF + k0, hi);
#pragma unroll
    for (int i = 0; i < 4; ++i)
#pragma unroll
      for (int j = 0; j < 2; ++j)
        acc[i][j] = __builtin_amdgcn_wmma_f32_16x16x32_f16(
            false, a[i], false, bm[j], (short)0, acc[i][j], false, false);
  }
  const float* sqb = sq + (size_t)b * NP;
#pragma unroll
  for (int j = 0; j < 2; ++j) {
    int nn = nbase + 16 * j + lo;          // D col = lane&15
    float sn = sqb[nn];
#pragma unroll
    for (int i = 0; i < 4; ++i) {
#pragma unroll
      for (int r = 0; r < 8; ++r) {
        int m = mbase + 16 * i + r + 8 * hi;  // D: vgpr r, lane-half hi -> row
        pd[((size_t)b * NP + m) * NP + nn] = 2.f * acc[i][j][r] - sqb[m] - sn;
      }
    }
  }
}

// ---------------------------------------------------------------- top-k (k=20) per row
__global__ void k_topk(const float* __restrict__ pd, int* __restrict__ idxO) {
  int t = blockIdx.x * blockDim.x + threadIdx.x;
  if (t >= BQ * NP) return;
  const float* row = pd + (size_t)t * NP;
  float best[KNN]; int bi[KNN];
#pragma unroll
  for (int i = 0; i < KNN; ++i) { best[i] = -3.4e38f; bi[i] = 0; }
  for (int m = 0; m < NP; m += 4) {
    float4 v4 = *(const float4*)(row + m);
    float vv[4] = { v4.x, v4.y, v4.z, v4.w };
#pragma unroll
    for (int q = 0; q < 4; ++q) {
      float v = vv[q];
      if (v > best[KNN - 1]) {
        int j = KNN - 1;
        while (j > 0 && best[j - 1] < v) {        // strict: stable ties
          best[j] = best[j - 1]; bi[j] = bi[j - 1]; --j;
        }
        best[j] = v; bi[j] = m + q;
      }
    }
  }
#pragma unroll
  for (int i = 0; i < KNN; ++i) idxO[(size_t)t * KNN + i] = bi[i];
}

// ---------------------------------------------------------------- fused edge conv
// One workgroup (256 thr, 8 waves) per point. Edge features (20 x 2C padded to
// 32 x TCP) staged f16 in LDS; 2 M-tiles x O/16 N-tiles of WMMA; BN + LeakyReLU
// + max-over-k fused on D fragments. Output stored f16 into concat buffer.
template <int TCP>
__global__ void k_edgeconv(const _Float16* __restrict__ h, int strideF, int C,
                           const int* __restrict__ idx,
                           const _Float16* __restrict__ w,   // (O, TCP) f16 padded
                           const float* __restrict__ bnp,    // (4, O)
                           _Float16* __restrict__ out, int outOff, int O) {
  __shared__ _Float16 feat[32 * TCP];
  __shared__ int      sidx[32];
  __shared__ _Float16 sctr[128];            // center features, C <= 128
  int p = blockIdx.x;
  int b = p / NP, n = p % NP;
  int twoC = 2 * C;
  const _Float16* hb = h + (size_t)b * NP * strideF;

  if (threadIdx.x < KNN) sidx[threadIdx.x] = idx[(size_t)p * KNN + threadIdx.x];
  for (int c = threadIdx.x; c < C; c += blockDim.x)
    sctr[c] = hb[(size_t)n * strideF + c];
  __syncthreads();

  for (int i = threadIdx.x; i < 32 * TCP; i += blockDim.x) {
    int e = i / TCP, c = i % TCP;
    float v = 0.f;
    if (e < KNN && c < twoC) {
      if (c < C) v = (float)hb[(size_t)sidx[e] * strideF + c] - (float)sctr[c];
      else       v = (float)sctr[c - C];
    }
    feat[i] = (_Float16)v;
  }
  __syncthreads();

  int wave = threadIdx.x >> 5, lane = threadIdx.x & 31;
  int lo = lane & 15, hi = lane >> 4;
  int nOT = O / 16;
  for (int ot = wave; ot < nOT; ot += 8) {
    const _Float16* wrow = w + (size_t)(ot * 16 + lo) * TCP;
    v8f acc0 = {}, acc1 = {};               // edge rows 0-15 and 16-31
#pragma unroll
    for (int k0 = 0; k0 < TCP; k0 += 32) {
      v16h a0 = ld_fragA(feat + lo * TCP + k0, hi);
      v16h a1 = ld_fragA(feat + (16 + lo) * TCP + k0, hi);
      v16h bw = ld_fragB(wrow + k0, hi);
      acc0 = __builtin_amdgcn_wmma_f32_16x16x32_f16(false, a0, false, bw,
                                                    (short)0, acc0, false, false);
      acc1 = __builtin_amdgcn_wmma_f32_16x16x32_f16(false, a1, false, bw,
                                                    (short)0, acc1, false, false);
    }
    int o = ot * 16 + lo;                   // D col = lane&15
    float s  = bnp[o] * rsqrtf(bnp[3 * O + o] + EPS);
    float tb = bnp[O + o] - bnp[2 * O + o] * s;
    float mx = -3.4e38f;
#pragma unroll
    for (int r = 0; r < 8; ++r) {
      int m1 = 16 + r + 8 * hi;             // rows 16..31: valid only < KNN
      mx = fmaxf(mx, lrelu(acc0[r] * s + tb));   // rows 0..15 always valid
      float z1 = lrelu(acc1[r] * s + tb);
      if (m1 < KNN) mx = fmaxf(mx, z1);
    }
    mx = fmaxf(mx, __shfl_xor(mx, 16, 32)); // merge lane halves (same col)
    if (lane < 16)
      out[(size_t)p * 512 + outOff + o] = (_Float16)mx;
  }
}

// ---------------------------------------------------------------- conv5 GEMM + BN + LReLU
// A = concat features (B*N, 512) f16; W = (1024, 512) f16; out h5 (B*N, 1024)
// f16. grid (M/128, 1024/128), block 256 (8 waves), 64x32 per wave (4x2 tiles).
__global__ void k_conv5(const _Float16* __restrict__ A,
                        const _Float16* __restrict__ W,
                        const float* __restrict__ bnp,
                        _Float16* __restrict__ Out) {
  constexpr int KD = 512, NOUT = 1024;
  int wave = threadIdx.x >> 5, lane = threadIdx.x & 31;
  int lo = lane & 15, hi = lane >> 4;
  int mbase = blockIdx.x * 128 + (wave & 1) * 64;
  int nbase = blockIdx.y * 128 + (wave >> 1) * 32;
  v8f acc[4][2] = {};
#pragma unroll 2
  for (int k0 = 0; k0 < KD; k0 += 32) {
    __builtin_prefetch(W + (size_t)(nbase + lo) * KD + k0 + 256, 0, 1);
    v16h a[4], bw[2];
#pragma unroll
    for (int i = 0; i < 4; ++i)
      a[i] = ld_fragA(A + (size_t)(mbase + 16 * i + lo) * KD + k0, hi);
#pragma unroll
    for (int j = 0; j < 2; ++j)
      bw[j] = ld_fragB(W + (size_t)(nbase + 16 * j + lo) * KD + k0, hi);
#pragma unroll
    for (int i = 0; i < 4; ++i)
#pragma unroll
      for (int j = 0; j < 2; ++j)
        acc[i][j] = __builtin_amdgcn_wmma_f32_16x16x32_f16(
            false, a[i], false, bw[j], (short)0, acc[i][j], false, false);
  }
#pragma unroll
  for (int j = 0; j < 2; ++j) {
    int o = nbase + 16 * j + lo;
    float s  = bnp[o] * rsqrtf(bnp[3 * NOUT + o] + EPS);
    float tb = bnp[NOUT + o] - bnp[2 * NOUT + o] * s;
#pragma unroll
    for (int i = 0; i < 4; ++i)
#pragma unroll
      for (int r = 0; r < 8; ++r) {
        int m = mbase + 16 * i + r + 8 * hi;
        Out[(size_t)m * NOUT + o] = (_Float16)lrelu(acc[i][j][r] * s + tb);
      }
  }
}

// ---------------------------------------------------------------- global max/mean pool
// grid (1024/256, B); p = concat[max(1024), mean(1024)] per batch, f32.
__global__ void k_pool(const _Float16* __restrict__ h5, float* __restrict__ p) {
  int b = blockIdx.y;
  int c = blockIdx.x * blockDim.x + threadIdx.x;
  float mx = -3.4e38f, sm = 0.f;
  for (int n = 0; n < NP; ++n) {
    float v = (float)h5[((size_t)b * NP + n) * 1024 + c];
    mx = fmaxf(mx, v);
    sm += v;
  }
  p[(size_t)b * 2048 + c]        = mx;
  p[(size_t)b * 2048 + 1024 + c] = sm * (1.f / NP);
}

// ---------------------------------------------------------------- tiny FC (M=16), f32
__global__ void k_fc(const float* __restrict__ in, const float* __restrict__ w,
                     const float* __restrict__ bnp, const float* __restrict__ bias,
                     float* __restrict__ out, int I, int O, int relu) {
  int t = blockIdx.x * blockDim.x + threadIdx.x;
  if (t >= BQ * O) return;
  int b = t / O, o = t % O;
  const float* row = in + (size_t)b * I;
  const float* wr  = w + (size_t)o * I;
  float acc = 0.f;
  for (int k = 0; k < I; ++k) acc += row[k] * wr[k];
  if (bnp) {
    float s = bnp[o] * rsqrtf(bnp[3 * O + o] + EPS);
    acc = acc * s + (bnp[O + o] - bnp[2 * O + o] * s);
  }
  if (bias) acc += bias[o];
  if (relu) acc = lrelu(acc);
  out[t] = acc;
}

// ================================================================ launcher
extern "C" void kernel_launch(void* const* d_in, const int* in_sizes, int n_in,
                              void* d_out, int out_size, void* d_ws, size_t ws_size,
                              hipStream_t stream) {
  const float* x   = (const float*)d_in[0];
  const float* w1  = (const float*)d_in[1];
  const float* bn1 = (const float*)d_in[2];
  const float* w2  = (const float*)d_in[3];
  const float* bn2 = (const float*)d_in[4];
  const float* w3  = (const float*)d_in[5];
  const float* bn3 = (const float*)d_in[6];
  const float* w4  = (const float*)d_in[7];
  const float* bn4 = (const float*)d_in[8];
  const float* w5  = (const float*)d_in[9];
  const float* bn5 = (const float*)d_in[10];
  const float* wl1 = (const float*)d_in[11];
  const float* bn6 = (const float*)d_in[12];
  const float* wl2 = (const float*)d_in[13];
  const float* bn7 = (const float*)d_in[14];
  const float* wl3 = (const float*)d_in[15];
  const float* bl3 = (const float*)d_in[16];
  // d_in[17] = k (device scalar); fixed at 20 per reference setup.
  (void)in_sizes; (void)n_in; (void)out_size; (void)ws_size;

  // Workspace carve (256B aligned), ~115 MB total.
  char* wsb = (char*)d_ws;
  size_t off = 0;
  auto carve = [&](size_t bytes) -> char* {
    char* q = wsb + off; off += (bytes + 255) & ~(size_t)255; return q;
  };
  _Float16* h0   = (_Float16*)carve((size_t)BQ * NP * 32 * 2);
  _Float16* xf   = (_Float16*)carve((size_t)BQ * NP * 512 * 2);   // concat [x1|x2|x3|x4]
  _Float16* h5   = (_Float16*)carve((size_t)BQ * NP * 1024 * 2);
  float*    pd   = (float*)   carve((size_t)BQ * NP * NP * 4);    // 64 MB, reused
  int*      idx  = (int*)     carve((size_t)BQ * NP * KNN * 4);
  float*    sq   = (float*)   carve((size_t)BQ * NP * 4);
  _Float16* w1p  = (_Float16*)carve((size_t)64  * 32  * 2);
  _Float16* w2p  = (_Float16*)carve((size_t)64  * 128 * 2);
  _Float16* w3p  = (_Float16*)carve((size_t)128 * 128 * 2);
  _Float16* w4p  = (_Float16*)carve((size_t)256 * 256 * 2);
  _Float16* w5p  = (_Float16*)carve((size_t)1024 * 512 * 2);
  float*    pool = (float*)   carve((size_t)BQ * 2048 * 4);
  float*    f1c  = (float*)   carve((size_t)BQ * 512 * 4);
  float*    f2c  = (float*)   carve((size_t)BQ * 256 * 4);

  const int TPB = 256;
  // Stage f16 inputs / padded weights.
  k_prep0<<<(BQ * NP + TPB - 1) / TPB, TPB, 0, stream>>>(x, h0);
  k_castw<<<(64  * 32  + TPB - 1) / TPB, TPB, 0, stream>>>(w1, w1p, 64, 6, 32);
  k_castw<<<(64  * 128 + TPB - 1) / TPB, TPB, 0, stream>>>(w2, w2p, 64, 128, 128);
  k_castw<<<(128 * 128 + TPB - 1) / TPB, TPB, 0, stream>>>(w3, w3p, 128, 128, 128);
  k_castw<<<(256 * 256 + TPB - 1) / TPB, TPB, 0, stream>>>(w4, w4p, 256, 256, 256);
  k_castw<<<(1024 * 512 + TPB - 1) / TPB, TPB, 0, stream>>>(w5, w5p, 1024, 512, 512);

  dim3 gpd(NP / 128, NP / 128, BQ);
  int gN = (BQ * NP + TPB - 1) / TPB;

  // ---- EdgeConv block 1: h0 (C=3, pad 32) -> xf[0:64)
  k_sqnorm<<<gN, TPB, 0, stream>>>(h0, 32, 32, sq);
  k_pd_wmma<32><<<gpd, TPB, 0, stream>>>(h0, 32, sq, pd);
  k_topk<<<gN, TPB, 0, stream>>>(pd, idx);
  k_edgeconv<32><<<BQ * NP, TPB, 0, stream>>>(h0, 32, 3, idx, w1p, bn1, xf, 0, 64);

  // ---- EdgeConv block 2: x1 (C=64) -> xf[64:128)
  k_sqnorm<<<gN, TPB, 0, stream>>>(xf + 0, 512, 64, sq);
  k_pd_wmma<64><<<gpd, TPB, 0, stream>>>(xf + 0, 512, sq, pd);
  k_topk<<<gN, TPB, 0, stream>>>(pd, idx);
  k_edgeconv<128><<<BQ * NP, TPB, 0, stream>>>(xf + 0, 512, 64, idx, w2p, bn2, xf, 64, 64);

  // ---- EdgeConv block 3: x2 (C=64) -> xf[128:256)
  k_sqnorm<<<gN, TPB, 0, stream>>>(xf + 64, 512, 64, sq);
  k_pd_wmma<64><<<gpd, TPB, 0, stream>>>(xf + 64, 512, sq, pd);
  k_topk<<<gN, TPB, 0, stream>>>(pd, idx);
  k_edgeconv<128><<<BQ * NP, TPB, 0, stream>>>(xf + 64, 512, 64, idx, w3p, bn3, xf, 128, 128);

  // ---- EdgeConv block 4: x3 (C=128) -> xf[256:512)
  k_sqnorm<<<gN, TPB, 0, stream>>>(xf + 128, 512, 128, sq);
  k_pd_wmma<128><<<gpd, TPB, 0, stream>>>(xf + 128, 512, sq, pd);
  k_topk<<<gN, TPB, 0, stream>>>(pd, idx);
  k_edgeconv<256><<<BQ * NP, TPB, 0, stream>>>(xf + 128, 512, 128, idx, w4p, bn4, xf, 256, 256);

  // ---- conv5 (B*N,512)x(512,1024) + BN + LReLU -> h5 f16
  dim3 g5((BQ * NP) / 128, 1024 / 128);
  k_conv5<<<g5, TPB, 0, stream>>>(xf, w5p, bn5, h5);

  // ---- global max/mean pool -> (B, 2048) f32
  dim3 gp(1024 / 256, BQ);
  k_pool<<<gp, TPB, 0, stream>>>(h5, pool);

  // ---- FC head (f32)
  k_fc<<<(BQ * 512 + TPB - 1) / TPB, TPB, 0, stream>>>(pool, wl1, bn6, nullptr, f1c, 2048, 512, 1);
  k_fc<<<(BQ * 256 + TPB - 1) / TPB, TPB, 0, stream>>>(f1c, wl2, bn7, nullptr, f2c, 512, 256, 1);
  k_fc<<<(BQ * 40  + TPB - 1) / TPB, TPB, 0, stream>>>(f2c, wl3, nullptr, bl3,
                                                       (float*)d_out, 256, 40, 0);
}
